// MultiHeadAttention_17592186045125
// MI455X (gfx1250) — matrix operand order repaired and checked by reading, not compiled, not run on hardware
//
#include <hip/hip_runtime.h>
#include <stdint.h>

typedef __bf16 bf16;
typedef __bf16 bf16x8  __attribute__((ext_vector_type(8)));
typedef __bf16 bf16x16 __attribute__((ext_vector_type(16)));
typedef float  f32x8   __attribute__((ext_vector_type(8)));

#define BM    128         // block tile M
#define KTILE 32
#define LDSP  40          // 80B row stride: 16B-aligned, conflict-free frag reads

// low 32 bits of a generic pointer to __shared__ data == LDS byte offset
__device__ __forceinline__ uint32_t lds_off32(const void* p) {
    return (uint32_t)(uintptr_t)p;
}
// CDNA5 async memory->LDS DMA (16B per lane), tracked by ASYNCcnt
__device__ __forceinline__ void async_b128(uint32_t lds, const bf16* g) {
    asm volatile("global_load_async_to_lds_b128 %0, %1, off"
                 :: "v"(lds), "v"(g) : "memory");
}
template<int N>
__device__ __forceinline__ void wait_async_le() {
    asm volatile("s_wait_asynccnt %0" :: "i"(N) : "memory");
}

// ---------------------------------------------------------------------------
// f32 -> bf16 conversion (grid-stride)
// ---------------------------------------------------------------------------
__global__ __launch_bounds__(256) void cvt_f32_bf16(const float* __restrict__ in,
                                                    bf16* __restrict__ out,
                                                    long long n) {
    long long i = (long long)blockIdx.x * blockDim.x + threadIdx.x;
    long long stride = (long long)gridDim.x * blockDim.x;
    for (; i < n; i += stride) out[i] = (bf16)in[i];
}

// ---------------------------------------------------------------------------
// fused f32 -> bf16 transpose: in [z][R][C] f32  ->  out [z][C][R] bf16
// ---------------------------------------------------------------------------
__global__ __launch_bounds__(256) void cvt_t_f32_bf16(const float* __restrict__ in,
                                                      bf16* __restrict__ out,
                                                      int R, int C) {
    __shared__ float t[32][33];
    long long z = blockIdx.z;
    in  += z * (long long)R * C;
    out += z * (long long)R * C;
    int bx = blockIdx.x * 32, by = blockIdx.y * 32;
    int tx = threadIdx.x, ty = threadIdx.y;          // block (32, 8)
#pragma unroll
    for (int i = 0; i < 32; i += 8)
        t[ty + i][tx] = in[(long long)(by + ty + i) * C + (bx + tx)];
    __syncthreads();
#pragma unroll
    for (int i = 0; i < 32; i += 8)
        out[(long long)(bx + ty + i) * R + (by + tx)] = (bf16)t[tx][ty + i];
}

// ---------------------------------------------------------------------------
// Batched GEMM: C = A[M,K] * Bt[N,K]^T ; A,Bt bf16 row-major (contiguous K).
// Block tile BM x BN, 256 threads (8 waves), double-buffered async LDS stage.
// Batch z -> (zb, zh) = (z / zdiv, z % zdiv) with 2-level strides.
// ---------------------------------------------------------------------------
struct GemmP {
    const bf16* A;  long long lda, sAb, sAh;
    const bf16* B;  long long ldb, sBb, sBh;
    void*       C;  long long ldc, sCb, sCh;
    int M, N, K, zdiv;
    int mode;           // 0: +biasN -> bf16         | 1: *scale +biasT, mask->0 -> bf16
                        // 2: raw f32                | 3: +biasN -> f32
                        // 4: +biasN -> bf16, write transposed per sub-batch (V proj)
    int mshift;         // mode 4: rows-per-batch = 1 << mshift
    float scale;
    const float* biasN;
    const float* biasT; long long sTb, sTh, ldT;
    const unsigned char* mask; long long sMb, ldM;
};

template<int BN>
__global__ __launch_bounds__(256) void gemm_bf16_wmma(GemmP p) {
    constexpr int NF = BN / 32;                  // N frags per wave (4 or 2)
    constexpr int NISSUE = (BN == 128) ? 4 : 3;  // async issues per thread per tile
    __shared__ bf16 As[2][BM][LDSP];
    __shared__ bf16 Bs[2][BN][LDSP];

    const int z  = blockIdx.z;
    const int zb = z / p.zdiv, zh = z % p.zdiv;
    const long long m0 = (long long)blockIdx.y * BM;
    const long long n0 = (long long)blockIdx.x * BN;
    const int tid = threadIdx.x;

    // ---- tile loader: thread -> (row lr, 16B chunk lc) ----
    const int lr = tid >> 2;                 // 0..63 (A also +64; B +64 iff BN=128)
    const int lc = tid & 3;                  // chunk of 8 bf16
    const bf16* aptr = p.A + zb * p.sAb + zh * p.sAh + (m0 + lr) * p.lda + lc * 8;
    const bf16* bptr = p.B + zb * p.sBb + zh * p.sBh + (n0 + lr) * p.ldb + lc * 8;
    const long long aRow64 = 64 * p.lda, bRow64 = 64 * p.ldb;
    const uint32_t aLds0 = lds_off32(&As[0][lr][lc * 8]);
    const uint32_t bLds0 = lds_off32(&Bs[0][lr][lc * 8]);
    const uint32_t aBufStride = (uint32_t)(BM * LDSP * 2);
    const uint32_t bBufStride = (uint32_t)(BN * LDSP * 2);
    const uint32_t ldsRow64 = 64 * LDSP * 2;

    auto issue_tile = [&](int buf, const bf16* ap, const bf16* bp) {
        uint32_t aL = aLds0 + buf * aBufStride;
        uint32_t bL = bLds0 + buf * bBufStride;
        async_b128(aL,            ap);
        async_b128(aL + ldsRow64, ap + aRow64);
        async_b128(bL,            bp);
        if (BN == 128) async_b128(bL + ldsRow64, bp + bRow64);
    };

    const int wave = tid >> 5, lane = tid & 31;
    const int wm = (wave >> 1) << 5;             // 0/32/64/96
    const int wn = (wave & 1) * (BN / 2);        // 0 or BN/2
    const int half = lane >> 4, l15 = lane & 15;

    f32x8 acc[2][NF] = {};

    // ---- double-buffered main loop ----
    issue_tile(0, aptr, bptr);
    aptr += KTILE; bptr += KTILE;
    int cur = 0;
    for (int k0 = 0; k0 < p.K; k0 += KTILE) {
        if (k0 + KTILE < p.K) {
            issue_tile(cur ^ 1, aptr, bptr);     // prefetch next tile
            aptr += KTILE; bptr += KTILE;
            wait_async_le<NISSUE>();             // in-order: tile `cur` complete
        } else {
            wait_async_le<0>();
        }
        __syncthreads();

        bf16x16 fa[2], fb[NF];
#pragma unroll
        for (int f = 0; f < 2; ++f) {
            // A frag (ISA 7.12.2): lane half -> K groups {8h..8h+7},{16+8h..}
            const bf16* pa = &As[cur][wm + f * 16 + l15][0];
            bf16x8 a0 = *(const bf16x8*)(pa + 8 * half);
            bf16x8 a1 = *(const bf16x8*)(pa + 16 + 8 * half);
            fa[f] = __builtin_shufflevector(a0, a1,
                    0,1,2,3,4,5,6,7,8,9,10,11,12,13,14,15);
        }
#pragma unroll
        for (int f = 0; f < NF; ++f) {
            // B frag: lane = column, lane half -> K groups {8h..8h+7},{16+8h..}
            const bf16* pb = &Bs[cur][wn + f * 16 + l15][0];
            bf16x8 b0 = *(const bf16x8*)(pb + 8 * half);
            bf16x8 b1 = *(const bf16x8*)(pb + 16 + 8 * half);
            fb[f] = __builtin_shufflevector(b0, b1,
                    0,1,2,3,4,5,6,7,8,9,10,11,12,13,14,15);
        }

#pragma unroll
        for (int fi = 0; fi < 2; ++fi)
#pragma unroll
            for (int fj = 0; fj < NF; ++fj)
                acc[fi][fj] = __builtin_amdgcn_wmma_f32_16x16x32_bf16(
                    false, fa[fi], false, fb[fj],
                    (short)0, acc[fi][fj], false, false);
        __syncthreads();
        cur ^= 1;
    }

    // ---- epilogue: C/D layout — VGPR j: M = j + 8*half, N = lane&15 ----
    const long long cOff = zb * p.sCb + zh * p.sCh;
#pragma unroll
    for (int fi = 0; fi < 2; ++fi)
#pragma unroll
    for (int fj = 0; fj < NF; ++fj)
#pragma unroll
    for (int j = 0; j < 8; ++j) {
        long long gm = m0 + wm + fi * 16 + 8 * half + j;
        long long gn = n0 + wn + fj * 16 + l15;
        float v = acc[fi][fj][j];
        if (p.mode == 0) {
            if (p.biasN) v += p.biasN[gn];
            ((bf16*)p.C)[cOff + gm * p.ldc + gn] = (bf16)v;
        } else if (p.mode == 1) {
            v = v * p.scale + p.biasT[zb * p.sTb + zh * p.sTh + gm * p.ldT + gn];
            if (p.mask[zb * p.sMb + gm * p.ldM + gn]) v = 0.0f;
            ((bf16*)p.C)[cOff + gm * p.ldc + gn] = (bf16)v;
        } else if (p.mode == 2) {
            ((float*)p.C)[cOff + gm * p.ldc + gn] = v;
        } else if (p.mode == 3) {
            ((float*)p.C)[cOff + gm * p.ldc + gn] = v + p.biasN[gn];
        } else {  // mode 4: split gm -> (batch, row); write C[b][gn][row]
            long long bb = gm >> p.mshift;
            long long rr = gm & ((1LL << p.mshift) - 1);
            if (p.biasN) v += p.biasN[gn];
            ((bf16*)p.C)[bb * p.sCb + gn * p.ldc + rr] = (bf16)v;
        }
    }
}

// ---------------------------------------------------------------------------
// Row softmax: f32 in -> bf16 probabilities out. One block per row.
// ---------------------------------------------------------------------------
__global__ __launch_bounds__(256) void softmax_rows(const float* __restrict__ X,
                                                    bf16* __restrict__ P,
                                                    int cols) {
    __shared__ float red[256];
    long long row = blockIdx.x;
    const float* x = X + row * (long long)cols;
    bf16* o = P + row * (long long)cols;
    int tid = threadIdx.x;

    float mx = -3.4e38f;
    for (int c = tid; c < cols; c += 256) mx = fmaxf(mx, x[c]);
    red[tid] = mx; __syncthreads();
    for (int s = 128; s > 0; s >>= 1) {
        if (tid < s) red[tid] = fmaxf(red[tid], red[tid + s]);
        __syncthreads();
    }
    mx = red[0]; __syncthreads();

    float sum = 0.0f;
    for (int c = tid; c < cols; c += 256) sum += __expf(x[c] - mx);
    red[tid] = sum; __syncthreads();
    for (int s = 128; s > 0; s >>= 1) {
        if (tid < s) red[tid] += red[tid + s];
        __syncthreads();
    }
    float inv = 1.0f / red[0];
    for (int c = tid; c < cols; c += 256)
        o[c] = (bf16)(__expf(x[c] - mx) * inv);
}

// ---------------------------------------------------------------------------
extern "C" void kernel_launch(void* const* d_in, const int* in_sizes, int n_in,
                              void* d_out, int out_size, void* d_ws, size_t ws_size,
                              hipStream_t stream) {
    const int B = 2, S = 2048, HID = 512, NH = 8, D = 64;
    const long long SS  = (long long)S * S;
    const long long BSH = (long long)B * S * HID;

    const float* q    = (const float*)d_in[0];
    const float* k    = (const float*)d_in[1];
    const float* v    = (const float*)d_in[2];
    const float* bias = (const float*)d_in[3];
    const unsigned char* mask = (const unsigned char*)d_in[4];
    const float* w    = (const float*)d_in[5];
    const float* Wq = (const float*)d_in[6];  const float* bq = (const float*)d_in[7];
    const float* Wk = (const float*)d_in[8];  const float* bk = (const float*)d_in[9];
    const float* Wv = (const float*)d_in[10]; const float* bv = (const float*)d_in[11];
    const float* Wo = (const float*)d_in[12]; const float* bo = (const float*)d_in[13];

    size_t off = 0;
    auto take = [&](size_t bytes) {
        char* r = (char*)d_ws + off;
        off += (bytes + 255) & ~(size_t)255;
        return (void*)r;
    };
    bf16* qbf  = (bf16*)take(BSH * 2);
    bf16* kbf  = (bf16*)take(BSH * 2);
    bf16* vbf  = (bf16*)take(BSH * 2);
    bf16* Wqb  = (bf16*)take((size_t)HID * HID * 2);
    bf16* Wkb  = (bf16*)take((size_t)HID * HID * 2);
    bf16* Wvb  = (bf16*)take((size_t)HID * HID * 2);
    bf16* Wob  = (bf16*)take((size_t)HID * HID * 2);
    bf16* wT   = (bf16*)take((size_t)B * SS * 2);        // w transposed [b][l][k]
    bf16* Qp   = (bf16*)take(BSH * 2);
    bf16* Kp   = (bf16*)take(BSH * 2);
    bf16* VpT  = (bf16*)take(BSH * 2);                   // V proj, [b][hid][s]
    bf16* Xbuf = (bf16*)take((size_t)B * NH * SS * 2);   // scores, reused for P
    float* X2  = (float*)take((size_t)B * NH * SS * 4);  // x @ w (f32)
    bf16* Obuf = (bf16*)take(BSH * 2);

    // ---- conversions ----
    cvt_f32_bf16<<<2048, 256, 0, stream>>>(q, qbf, BSH);
    cvt_f32_bf16<<<2048, 256, 0, stream>>>(k, kbf, BSH);
    cvt_f32_bf16<<<2048, 256, 0, stream>>>(v, vbf, BSH);
    cvt_f32_bf16<<<512, 256, 0, stream>>>(Wq, Wqb, (long long)HID * HID);
    cvt_f32_bf16<<<512, 256, 0, stream>>>(Wk, Wkb, (long long)HID * HID);
    cvt_f32_bf16<<<512, 256, 0, stream>>>(Wv, Wvb, (long long)HID * HID);
    cvt_f32_bf16<<<512, 256, 0, stream>>>(Wo, Wob, (long long)HID * HID);
    // w[b][k][l] -> wT[b][l][k] (bf16), so x@w reads B contiguously
    cvt_t_f32_bf16<<<dim3(S / 32, S / 32, B), dim3(32, 8), 0, stream>>>(w, wT, S, S);

    GemmP p{};

    // ---- Q/K/V projections: [4096,512] x [512,512]^T + bias -> bf16 ----
    auto proj = [&](const bf16* Ain, const bf16* Win, const float* bvec,
                    bf16* Cout, int mode) {
        GemmP g{};
        g.A = Ain; g.lda = HID;
        g.B = Win; g.ldb = HID;
        g.C = Cout;
        g.M = B * S; g.N = HID; g.K = HID; g.zdiv = 1;
        g.mode = mode; g.biasN = bvec;
        if (mode == 4) { g.ldc = S; g.sCb = (long long)HID * S; g.mshift = 11; }
        else           { g.ldc = HID; }
        gemm_bf16_wmma<128><<<dim3(HID / 128, (B * S) / BM, 1), 256, 0, stream>>>(g);
    };
    proj(qbf, Wqb, bq, Qp, 0);
    proj(kbf, Wkb, bk, Kp, 0);
    proj(vbf, Wvb, bv, VpT, 4);   // V written transposed: [b][hid][s]

    // ---- scores: x[b,h,q,k] = scale*Qh.Kh^T + attn_bias; mask -> 0 ----
    p = GemmP{};
    p.A = Qp;  p.lda = HID; p.sAb = (long long)S * HID; p.sAh = D;
    p.B = Kp;  p.ldb = HID; p.sBb = (long long)S * HID; p.sBh = D;
    p.C = Xbuf; p.ldc = S;  p.sCb = (long long)NH * SS; p.sCh = SS;
    p.M = S; p.N = S; p.K = D; p.zdiv = NH;
    p.mode = 1; p.scale = 0.125f;  // D^-0.5
    p.biasT = bias; p.sTb = (long long)NH * SS; p.sTh = SS; p.ldT = S;
    p.mask = mask;  p.sMb = SS; p.ldM = S;
    gemm_bf16_wmma<128><<<dim3(S / 128, S / BM, B * NH), 256, 0, stream>>>(p);

    // ---- x2[b,h,q,l] = x @ w  (B = wT, contiguous) -> f32 ----
    p = GemmP{};
    p.A = Xbuf; p.lda = S; p.sAb = (long long)NH * SS; p.sAh = SS;
    p.B = wT;   p.ldb = S; p.sBb = SS; p.sBh = 0;
    p.C = X2;   p.ldc = S; p.sCb = (long long)NH * SS; p.sCh = SS;
    p.M = S; p.N = S; p.K = S; p.zdiv = NH;
    p.mode = 2;
    gemm_bf16_wmma<128><<<dim3(S / 128, S / BM, B * NH), 256, 0, stream>>>(p);

    // ---- softmax rows -> bf16 P (reuse Xbuf) ----
    softmax_rows<<<B * NH * S, 256, 0, stream>>>(X2, Xbuf, S);

    // ---- o = P @ Vh  (B = VpT rows, contiguous) -> Obuf[b,s,hid] ----
    p = GemmP{};
    p.A = Xbuf; p.lda = S;   p.sAb = (long long)NH * SS; p.sAh = SS;
    p.B = VpT;  p.ldb = S;   p.sBb = (long long)HID * S; p.sBh = (long long)D * S;
    p.C = Obuf; p.ldc = HID; p.sCb = (long long)S * HID; p.sCh = D;
    p.M = S; p.N = D; p.K = S; p.zdiv = NH;
    p.mode = 0; p.biasN = nullptr;
    gemm_bf16_wmma<64><<<dim3(D / 64, S / BM, B * NH), 256, 0, stream>>>(p);

    // ---- out = Obuf @ Wo^T + bo -> f32 d_out ----
    p = GemmP{};
    p.A = Obuf;  p.lda = HID;
    p.B = Wob;   p.ldb = HID;
    p.C = d_out; p.ldc = HID;
    p.M = B * S; p.N = HID; p.K = HID; p.zdiv = 1;
    p.mode = 3; p.biasN = bo;
    gemm_bf16_wmma<128><<<dim3(HID / 128, (B * S) / BM, 1), 256, 0, stream>>>(p);
}
